// CAM_49495203119665
// MI455X (gfx1250) — matrix operand order repaired and checked by reading, not compiled, not run on hardware
//
#include <hip/hip_runtime.h>

// CAM (channel attention) for inputs [B=16, H=64, W=64, C=512] fp32.
//   aTa  = A^T A  (per batch, A = [4096, 512])  -- symmetric: lower blocks only
//   attn = softmax_rows(aTa)
//   out  = gamma * (A attn^T) + inputs
// GEMMs on v_wmma_f32_16x16x32_f16 with double-buffered LDS staging.

typedef __attribute__((ext_vector_type(16))) _Float16 v16h;
typedef __attribute__((ext_vector_type(8)))  float    v8f;

#define B_   16
#define N_   4096
#define C_   512
#define LROW 40   // LDS row stride in halves: 32 data + 8 pad = 80 B (16B aligned)

__device__ __forceinline__ unsigned pack2h(float a, float b) {
  union { _Float16 h[2]; unsigned u; } p;
  p.h[0] = (_Float16)a; p.h[1] = (_Float16)b;
  return p.u;
}

// ---------------------------------------------------------------------------
// Kernel 1: Gram matrix  aTa[b,i,j] = sum_n A[n,i]*A[n,j]   (symmetric)
// Grid (4,4,16), block 256. WG tile 128i x 128j, wave tile 64i x 32j.
// Blocks with j-tile > i-tile exit; blocks with j-tile < i-tile mirror writes.
// ---------------------------------------------------------------------------
struct StageG { float4 x0[4], x1[4]; };

__device__ __forceinline__ void gram_gload(const float* __restrict__ A,
                                           int i0, int j0, int k0, int t, StageG& r) {
  #pragma unroll
  for (int u = 0; u < 4; ++u) {
    int linear = t + 256 * u;            // 0..1023
    int blk = linear >> 9;               // 0 = i-block, 1 = j-block
    int rem = linear & 511;
    int np  = rem >> 5;                  // n-pair 0..15
    int c4  = rem & 31;                  // 4-channel chunk 0..31
    int ch0 = blk ? j0 : i0;
    int n   = np * 2;
    r.x0[u] = *(const float4*)(A + (size_t)(k0 + n    ) * C_ + ch0 + c4 * 4);
    r.x1[u] = *(const float4*)(A + (size_t)(k0 + n + 1) * C_ + ch0 + c4 * 4);
  }
}

__device__ __forceinline__ void gram_sstore(_Float16* buf, int t, const StageG& r) {
  #pragma unroll
  for (int u = 0; u < 4; ++u) {
    int linear = t + 256 * u;
    int blk = linear >> 9;
    int rem = linear & 511;
    int np  = rem >> 5;
    int c4  = rem & 31;
    int chb = blk * 128 + c4 * 4;
    unsigned* dw = (unsigned*)buf;       // dword index = 20*ch + n/2
    const float* p0 = (const float*)&r.x0[u];
    const float* p1 = (const float*)&r.x1[u];
    #pragma unroll
    for (int e = 0; e < 4; ++e)
      dw[20 * (chb + e) + np] = pack2h(p0[e], p1[e]);
  }
}

__global__ __launch_bounds__(256)
void cam_gram(const float* __restrict__ in, float* __restrict__ aTa) {
  __shared__ _Float16 lds[2 * 256 * LROW];   // double buffer, 40 KB

  const int i0 = blockIdx.y * 128;
  const int j0 = blockIdx.x * 128;
  if (j0 > i0) return;                       // symmetric: lower blocks only
  const bool mirror = (j0 < i0);

  const int b  = blockIdx.z;
  const int t  = threadIdx.x;
  const int lane  = t & 31;
  const int wave  = t >> 5;
  const int wi    = wave >> 2;               // 0..1 -> 64 i rows
  const int wj    = wave & 3;                // 0..3 -> 32 j cols
  const int lrow  = lane & 15;
  const int khalf = (lane >> 4) * 8;         // A-frag split-K offset (halves)
  const int bbase = (lane >> 4) * 16;        // unused; B-frag reads use khalf (same split-K as A)

  const float* A = in + (size_t)b * N_ * C_;

  v8f acc[4][2];
  #pragma unroll
  for (int m = 0; m < 4; ++m)
    #pragma unroll
    for (int s = 0; s < 2; ++s) acc[m][s] = (v8f){0.f,0.f,0.f,0.f,0.f,0.f,0.f,0.f};

  StageG sr;
  gram_gload(A, i0, j0, 0, t, sr);
  gram_sstore(lds, t, sr);
  __syncthreads();

  for (int s = 0; s < N_ / 32; ++s) {
    _Float16* cur = lds + (s & 1) * (256 * LROW);

    // Issue fragment ds reads first (DS completes in order, so WMMA's
    // dscnt wait does not cover the later staging stores).
    v16h afrag[4], bfrag[2];
    #pragma unroll
    for (int m = 0; m < 4; ++m) {
      const _Float16* rp = cur + (wi * 64 + m * 16 + lrow) * LROW;
      uint4* f = (uint4*)&afrag[m];
      f[0] = *(const uint4*)(rp + khalf);
      f[1] = *(const uint4*)(rp + 16 + khalf);
    }
    #pragma unroll
    for (int q = 0; q < 2; ++q) {
      const _Float16* rp = cur + (128 + wj * 32 + q * 16 + lrow) * LROW;
      uint4* f = (uint4*)&bfrag[q];
      f[0] = *(const uint4*)(rp + khalf);
      f[1] = *(const uint4*)(rp + 16 + khalf);
    }

    // Prefetch next tile into registers while WMMAs run.
    if (s + 1 < N_ / 32) gram_gload(A, i0, j0, (s + 1) * 32, t, sr);

    #pragma unroll
    for (int m = 0; m < 4; ++m)
      #pragma unroll
      for (int q = 0; q < 2; ++q)
        acc[m][q] = __builtin_amdgcn_wmma_f32_16x16x32_f16(
            false, afrag[m], false, bfrag[q], (short)0, acc[m][q], false, false);

    if (s + 1 < N_ / 32)
      gram_sstore(lds + ((s + 1) & 1) * (256 * LROW), t, sr);
    __syncthreads();
  }

  // C/D layout: VGPR r -> row r (lanes 0-15) or r+8 (lanes 16-31); lane -> col.
  const int col  = lane & 15;
  const int roff = (lane >> 4) * 8;
  #pragma unroll
  for (int m = 0; m < 4; ++m)
    #pragma unroll
    for (int q = 0; q < 2; ++q)
      #pragma unroll
      for (int r = 0; r < 8; ++r) {
        int gi = i0 + wi * 64 + m * 16 + roff + r;
        int gj = j0 + wj * 32 + q * 16 + col;
        float v = acc[m][q][r];
        aTa[((size_t)b * C_ + gi) * C_ + gj] = v;
        if (mirror) aTa[((size_t)b * C_ + gj) * C_ + gi] = v;
      }
}

// ---------------------------------------------------------------------------
// Kernel 2: row softmax of aTa -> attn (f16). One 256-thread block per row.
// ---------------------------------------------------------------------------
__global__ __launch_bounds__(256)
void cam_softmax(const float* __restrict__ aTa, _Float16* __restrict__ attn) {
  __shared__ float red[8];
  const int row = blockIdx.x;
  const int t   = threadIdx.x;
  const float2 v = ((const float2*)(aTa + (size_t)row * C_))[t];

  float m = fmaxf(v.x, v.y);
  #pragma unroll
  for (int off = 16; off; off >>= 1) m = fmaxf(m, __shfl_xor(m, off, 32));
  if ((t & 31) == 0) red[t >> 5] = m;
  __syncthreads();
  float M = red[0];
  #pragma unroll
  for (int w = 1; w < 8; ++w) M = fmaxf(M, red[w]);

  const float e0 = __expf(v.x - M);
  const float e1 = __expf(v.y - M);
  float s = e0 + e1;
  #pragma unroll
  for (int off = 16; off; off >>= 1) s += __shfl_xor(s, off, 32);
  __syncthreads();
  if ((t & 31) == 0) red[t >> 5] = s;
  __syncthreads();
  float S = 0.f;
  #pragma unroll
  for (int w = 0; w < 8; ++w) S += red[w];

  const float inv = 1.f / S;
  ((unsigned*)(attn + (size_t)row * C_))[t] = pack2h(e0 * inv, e1 * inv);
}

// ---------------------------------------------------------------------------
// Kernel 3: out[b,n,i] = gamma * sum_j attn[i,j]*A[n,j] + in[b,n,i]
// Grid (4,32,16), block 256. WG tile 128n x 128i, wave tile 64n x 32i.
// ---------------------------------------------------------------------------
struct StageO { float4 x[4]; };

__device__ __forceinline__ void out_gload(const float* __restrict__ A,
                                          int n0, int k0, int t, StageO& r) {
  #pragma unroll
  for (int u = 0; u < 4; ++u) {
    int linear = t + 256 * u;            // 0..1023
    int n  = linear >> 3;                // 0..127
    int j4 = linear & 7;                 // 0..7
    r.x[u] = *(const float4*)(A + (size_t)(n0 + n) * C_ + k0 + j4 * 4);
  }
}

__device__ __forceinline__ void out_sstore(_Float16* buf, int t, const StageO& r) {
  #pragma unroll
  for (int u = 0; u < 4; ++u) {
    int linear = t + 256 * u;
    int n  = linear >> 3;
    int j4 = linear & 7;
    unsigned* dw = (unsigned*)buf + 20 * n + 2 * j4;
    dw[0] = pack2h(r.x[u].x, r.x[u].y);
    dw[1] = pack2h(r.x[u].z, r.x[u].w);
  }
}

__global__ __launch_bounds__(256)
void cam_out(const float* __restrict__ in, const _Float16* __restrict__ attn,
             const float* __restrict__ gamma, float* __restrict__ out) {
  __shared__ _Float16 lds[2 * 128 * LROW];   // double buffer, 20 KB

  const int b  = blockIdx.z;
  const int n0 = blockIdx.y * 128;
  const int i0 = blockIdx.x * 128;
  const int t  = threadIdx.x;
  const int lane  = t & 31;
  const int wave  = t >> 5;
  const int wn    = wave >> 2;               // 0..1 -> 64 n rows
  const int wc    = wave & 3;                // 0..3 -> 32 i cols
  const int lrow  = lane & 15;
  const int khalf = (lane >> 4) * 8;
  const int bbase = (lane >> 4) * 16;

  const float*    A  = in   + (size_t)b * N_ * C_;
  const _Float16* At = attn + (size_t)b * C_ * C_;
  const float g = gamma[0];

  v8f acc[4][2];
  #pragma unroll
  for (int m = 0; m < 4; ++m)
    #pragma unroll
    for (int s = 0; s < 2; ++s) acc[m][s] = (v8f){0.f,0.f,0.f,0.f,0.f,0.f,0.f,0.f};

  StageO sr;
  out_gload(A, n0, 0, t, sr);
  out_sstore(lds, t, sr);
  __syncthreads();

  for (int s = 0; s < C_ / 32; ++s) {
    const int k0 = s * 32;
    _Float16* cur = lds + (s & 1) * (128 * LROW);

    v16h afrag[4], bfrag[2];
    #pragma unroll
    for (int m = 0; m < 4; ++m) {
      const _Float16* rp = cur + (wn * 64 + m * 16 + lrow) * LROW;
      uint4* f = (uint4*)&afrag[m];
      f[0] = *(const uint4*)(rp + khalf);
      f[1] = *(const uint4*)(rp + 16 + khalf);
    }
    #pragma unroll
    for (int q = 0; q < 2; ++q) {          // B column = attn row i (global f16)
      int colI = i0 + wc * 32 + q * 16 + lrow;
      const _Float16* rp = At + (size_t)colI * C_ + k0 + khalf;
      uint4* f = (uint4*)&bfrag[q];
      f[0] = *(const uint4*)(rp);
      f[1] = *(const uint4*)(rp + 16);
    }

    if (s + 1 < C_ / 32) out_gload(A, n0, (s + 1) * 32, t, sr);

    #pragma unroll
    for (int m = 0; m < 4; ++m)
      #pragma unroll
      for (int q = 0; q < 2; ++q)
        acc[m][q] = __builtin_amdgcn_wmma_f32_16x16x32_f16(
            false, afrag[m], false, bfrag[q], (short)0, acc[m][q], false, false);

    if (s + 1 < C_ / 32)
      out_sstore(lds + ((s + 1) & 1) * (128 * LROW), t, sr);
    __syncthreads();
  }

  const int col  = lane & 15;
  const int roff = (lane >> 4) * 8;
  #pragma unroll
  for (int m = 0; m < 4; ++m)
    #pragma unroll
    for (int q = 0; q < 2; ++q)
      #pragma unroll
      for (int r = 0; r < 8; ++r) {
        int n = n0 + wn * 64 + m * 16 + roff + r;
        int i = i0 + wc * 32 + q * 16 + col;
        size_t rel = (size_t)n * C_ + i;
        out[(size_t)b * N_ * C_ + rel] = g * acc[m][q][r] + A[rel];
      }
}

// ---------------------------------------------------------------------------
extern "C" void kernel_launch(void* const* d_in, const int* in_sizes, int n_in,
                              void* d_out, int out_size, void* d_ws, size_t ws_size,
                              hipStream_t stream) {
  const float* in    = (const float*)d_in[0];
  const float* gamma = (const float*)d_in[1];
  float* out = (float*)d_out;

  // workspace: aTa (16 MB f32) then attn (8 MB f16)
  float*    aTa  = (float*)d_ws;
  _Float16* attn = (_Float16*)((char*)d_ws + (size_t)B_ * C_ * C_ * sizeof(float));

  cam_gram   <<<dim3(C_ / 128, C_ / 128, B_), 256, 0, stream>>>(in, aTa);
  cam_softmax<<<dim3(B_ * C_),                256, 0, stream>>>(aTa, attn);
  cam_out    <<<dim3(C_ / 128, N_ / 128, B_), 256, 0, stream>>>(in, attn, gamma, out);
}
